// SpatialTransformer_47416438948536
// MI455X (gfx1250) — matrix operand order repaired and checked by reading, not compiled, not run on hardware
//
#include <hip/hip_runtime.h>
#include <hip/hip_bf16.h>

typedef __attribute__((ext_vector_type(16))) _Float16 v16h;
typedef __attribute__((ext_vector_type(8)))  _Float16 v8h;
typedef __attribute__((ext_vector_type(8)))  float    v8f;

#define BATCH 16
#define FEAT  (32*32*31)      // 31744
#define KSTEPS (FEAT/32)      // 992
#define KCHUNKS 31            // 992 / 31 = 32 wmma steps per chunk
#define STEPS_PER_CHUNK (KSTEPS / KCHUNKS)
#define BN_EPS 1e-3f
#define NEG_SLOPE 0.1f

// ---------------- conv1 (1->8) + relu + maxpool2 : 256x256 -> 128x128x8 ----
__global__ __launch_bounds__(256)
void conv1_pool(const float* __restrict__ x, const float* __restrict__ k,
                const float* __restrict__ bias, float* __restrict__ out) {
    int idx = blockIdx.x * blockDim.x + threadIdx.x;
    if (idx >= BATCH * 128 * 128) return;
    int px = idx & 127, py = (idx >> 7) & 127, b = idx >> 14;
    float best[8];
#pragma unroll
    for (int oc = 0; oc < 8; ++oc) best[oc] = 0.f;   // relu output >= 0
#pragma unroll
    for (int dy = 0; dy < 2; ++dy)
#pragma unroll
    for (int dx = 0; dx < 2; ++dx) {
        int cy = py * 2 + dy, cx = px * 2 + dx;
        float acc[8];
#pragma unroll
        for (int oc = 0; oc < 8; ++oc) acc[oc] = 0.f;
#pragma unroll
        for (int ky = 0; ky < 3; ++ky) {
            int iy = cy + ky - 1;
            if (iy < 0 || iy >= 256) continue;
#pragma unroll
            for (int kx = 0; kx < 3; ++kx) {
                int ix = cx + kx - 1;
                if (ix < 0 || ix >= 256) continue;
                float v = x[((size_t)b * 256 + iy) * 256 + ix];
                const float* kk = k + (ky * 3 + kx) * 8;
#pragma unroll
                for (int oc = 0; oc < 8; ++oc) acc[oc] += v * kk[oc];
            }
        }
#pragma unroll
        for (int oc = 0; oc < 8; ++oc) {
            float r = acc[oc] + bias[oc];
            r = r > 0.f ? r : 0.f;
            best[oc] = fmaxf(best[oc], r);
        }
    }
#pragma unroll
    for (int oc = 0; oc < 8; ++oc) out[(size_t)idx * 8 + oc] = best[oc];
}

// ---------------- conv2 (8->16) + relu + maxpool2 : 128x128 -> 64x64x16 ----
__global__ __launch_bounds__(256)
void conv2_pool(const float* __restrict__ in, const float* __restrict__ k,
                const float* __restrict__ bias, float* __restrict__ out) {
    int idx = blockIdx.x * blockDim.x + threadIdx.x;
    if (idx >= BATCH * 64 * 64) return;
    int px = idx & 63, py = (idx >> 6) & 63, b = idx >> 12;
    float best[16];
#pragma unroll
    for (int oc = 0; oc < 16; ++oc) best[oc] = 0.f;
#pragma unroll
    for (int dy = 0; dy < 2; ++dy)
#pragma unroll
    for (int dx = 0; dx < 2; ++dx) {
        int cy = py * 2 + dy, cx = px * 2 + dx;
        float acc[16];
#pragma unroll
        for (int oc = 0; oc < 16; ++oc) acc[oc] = 0.f;
        for (int ky = 0; ky < 3; ++ky) {
            int iy = cy + ky - 1;
            if (iy < 0 || iy >= 128) continue;
            for (int kx = 0; kx < 3; ++kx) {
                int ix = cx + kx - 1;
                if (ix < 0 || ix >= 128) continue;
                const float* pv = in + (((size_t)b * 128 + iy) * 128 + ix) * 8;
                const float* kt = k + (ky * 3 + kx) * 8 * 16;
#pragma unroll
                for (int ic = 0; ic < 8; ++ic) {
                    float v = pv[ic];
                    const float* kk = kt + ic * 16;
#pragma unroll
                    for (int oc = 0; oc < 16; ++oc) acc[oc] += v * kk[oc];
                }
            }
        }
#pragma unroll
        for (int oc = 0; oc < 16; ++oc) {
            float r = acc[oc] + bias[oc];
            r = r > 0.f ? r : 0.f;
            best[oc] = fmaxf(best[oc], r);
        }
    }
#pragma unroll
    for (int oc = 0; oc < 16; ++oc) out[(size_t)idx * 16 + oc] = best[oc];
}

// ---------------- conv3 (16->31) + relu + maxpool2 : 64x64 -> 32x32x31 -----
__global__ __launch_bounds__(256)
void conv3_pool(const float* __restrict__ in, const float* __restrict__ k,
                const float* __restrict__ bias, float* __restrict__ out) {
    int idx = blockIdx.x * blockDim.x + threadIdx.x;
    if (idx >= BATCH * 32 * 32) return;
    int px = idx & 31, py = (idx >> 5) & 31, b = idx >> 10;
    float best[31];
#pragma unroll
    for (int oc = 0; oc < 31; ++oc) best[oc] = 0.f;
    for (int dy = 0; dy < 2; ++dy)
    for (int dx = 0; dx < 2; ++dx) {
        int cy = py * 2 + dy, cx = px * 2 + dx;
        float acc[31];
#pragma unroll
        for (int oc = 0; oc < 31; ++oc) acc[oc] = 0.f;
        for (int ky = 0; ky < 3; ++ky) {
            int iy = cy + ky - 1;
            if (iy < 0 || iy >= 64) continue;
            for (int kx = 0; kx < 3; ++kx) {
                int ix = cx + kx - 1;
                if (ix < 0 || ix >= 64) continue;
                const float* pv = in + (((size_t)b * 64 + iy) * 64 + ix) * 16;
                const float* kt = k + (ky * 3 + kx) * 16 * 31;
                for (int ic = 0; ic < 16; ++ic) {
                    float v = pv[ic];
                    const float* kk = kt + ic * 31;
#pragma unroll
                    for (int oc = 0; oc < 31; ++oc) acc[oc] += v * kk[oc];
                }
            }
        }
#pragma unroll
        for (int oc = 0; oc < 31; ++oc) {
            float r = acc[oc] + bias[oc];
            r = r > 0.f ? r : 0.f;
            best[oc] = fmaxf(best[oc], r);
        }
    }
#pragma unroll
    for (int oc = 0; oc < 31; ++oc) out[(size_t)idx * 31 + oc] = best[oc];
}

// ---------------- BN + cast to f16 features -------------------------------
__global__ __launch_bounds__(256)
void bn_cast(const float* __restrict__ xi, const float* __restrict__ gamma,
             const float* __restrict__ beta, const float* __restrict__ mean,
             const float* __restrict__ var, _Float16* __restrict__ fh) {
    int i = blockIdx.x * blockDim.x + threadIdx.x;
    if (i >= BATCH * FEAT) return;
    int j = i % FEAT;
    float v = (xi[i] - mean[j]) * rsqrtf(var[j] + BN_EPS) * gamma[j] + beta[j];
    fh[i] = (_Float16)v;
}

// ---- pack d1_k (FEAT x 64, f32) into per-lane WMMA B-fragment order (f16) -
// Contiguous k order used here, same as A in dense1 (B, 32x16): lane L: n=L&15, hi=L>>4;
//   VGPR v holds K = 2v + hi*16 (+1). Packed so each lane reads 16 contiguous halves.
__global__ __launch_bounds__(256)
void pack_w1(const float* __restrict__ d1_k, _Float16* __restrict__ wp) {
    int idx = blockIdx.x * blockDim.x + threadIdx.x;   // (t, s, lane)
    if (idx >= 4 * KSTEPS * 32) return;
    int lane = idx & 31;
    int s = (idx >> 5) % KSTEPS;
    int t = idx / (32 * KSTEPS);
    int n = lane & 15, hi = lane >> 4;
    _Float16* dst = wp + (size_t)idx * 16;
    int kb = s * 32 + hi * 16;
#pragma unroll
    for (int v = 0; v < 8; ++v) {
        int k0 = kb + 2 * v;
        dst[2 * v]     = (_Float16)d1_k[(size_t)k0 * 64 + t * 16 + n];
        dst[2 * v + 1] = (_Float16)d1_k[(size_t)(k0 + 1) * 64 + t * 16 + n];
    }
}

// ---------------- dense1 via v_wmma_f32_16x16x32_f16, K-split -------------
// M=16 (batch), N=64 (4 tiles), K=31744 split into 31 chunks of 32 steps.
// Grid: (KCHUNKS, 4), one wave per block -> 124 waves spread across WGPs so
// the 4 MB weight stream is pulled through many L2 paths in parallel.
// Partials land in f1p[kc][16][64]; reduced (+bias+relu) in dense23.
__global__ __launch_bounds__(32)
void dense1_wmma(const _Float16* __restrict__ fh, const _Float16* __restrict__ wp,
                 float* __restrict__ f1p) {
    const int lane = threadIdx.x & 31;
    const int kc   = blockIdx.x;             // k-chunk 0..30
    const int t    = blockIdx.y;             // n-tile 0..3
    const int m    = lane & 15;
    const int hi   = lane >> 4;
    v8f c = {};
    const int s0 = kc * STEPS_PER_CHUNK;
    // A fragment (16-bit 16x32): lane covers K = hi*16 + [0..15]
    const _Float16* aRow  = fh + (size_t)m * FEAT + hi * 16 + (size_t)s0 * 32;
    const _Float16* bBase = wp + ((size_t)t * KSTEPS * 32 + (size_t)s0 * 32 + lane) * 16;
    for (int s = 0; s < STEPS_PER_CHUNK; ++s) {
        v8h alo = *(const v8h*)(aRow + (size_t)s * 32);
        v8h ahi = *(const v8h*)(aRow + (size_t)s * 32 + 8);
        v16h a = __builtin_shufflevector(alo, ahi,
                 0,1,2,3,4,5,6,7,8,9,10,11,12,13,14,15);
        v8h blo = *(const v8h*)(bBase + (size_t)s * 32 * 16);
        v8h bhi = *(const v8h*)(bBase + (size_t)s * 32 * 16 + 8);
        v16h b = __builtin_shufflevector(blo, bhi,
                 0,1,2,3,4,5,6,7,8,9,10,11,12,13,14,15);
        c = __builtin_amdgcn_wmma_f32_16x16x32_f16(
                false, a, false, b, (short)0, c, false, false);
    }
    // D layout: VGPR r, lanes 0-15: M=r, N=lane; lanes 16-31: M=8+r, N=lane-16
    int n = lane & 15;
#pragma unroll
    for (int r = 0; r < 8; ++r) {
        int mm = hi * 8 + r;
        f1p[((size_t)kc * 16 + mm) * 64 + t * 16 + n] = c[r];
    }
}

// -------- reduce K-partials + bias + relu, dense2 (64->96), dense3 --------
__global__ __launch_bounds__(96)
void dense23(const float* __restrict__ f1p, const float* __restrict__ d1b,
             const float* __restrict__ d2k, const float* __restrict__ d2b,
             const float* __restrict__ d3k, const float* __restrict__ d3b,
             float* __restrict__ theta) {
    __shared__ float s1[64];
    __shared__ float s2[96];
    int b = blockIdx.x, j = threadIdx.x;
    if (j < 64) {
        float acc = d1b[j];
        for (int kc = 0; kc < KCHUNKS; ++kc)
            acc += f1p[((size_t)kc * 16 + b) * 64 + j];
        s1[j] = acc > 0.f ? acc : 0.f;
    }
    __syncthreads();
    float acc = d2b[j];
    for (int k2 = 0; k2 < 64; ++k2) acc += s1[k2] * d2k[k2 * 96 + j];
    s2[j] = acc > 0.f ? acc : 0.f;
    __syncthreads();
    if (j < 6) {
        float th = d3b[j];
        for (int k2 = 0; k2 < 96; ++k2) th += s2[k2] * d3k[k2 * 6 + j];
        theta[b * 6 + j] = th;
    }
}

// ------- fused: affine grid + nearest-upsample + concat + bilinear + leaky -
// 8 threads per output pixel, 4 channels each -> coalesced float4 stores.
__global__ __launch_bounds__(256)
void sampler(const float* __restrict__ x, const float* __restrict__ xi,
             const float* __restrict__ theta, float* __restrict__ out) {
    int idx = blockIdx.x * blockDim.x + threadIdx.x;
    if (idx >= BATCH * 256 * 256 * 8) return;
    int g = idx & 7;
    int t = idx >> 3;
    int X = t & 255; t >>= 8;
    int Y = t & 255;
    int b = t >> 8;
    const float* th = theta + b * 6;
    float gx = X * (2.f / 255.f) - 1.f;
    float gy = Y * (2.f / 255.f) - 1.f;
    float u = gx * th[0] + gy * th[3] + th[2];
    float v = gx * th[1] + gy * th[4] + th[5];
    float xf = 0.5f * (u + 1.f) * 255.f;
    float yf = 0.5f * (v + 1.f) * 255.f;
    float x0f = floorf(xf), y0f = floorf(yf);
    float x1f = x0f + 1.f, y1f = y0f + 1.f;
    x0f = fminf(fmaxf(x0f, 0.f), 255.f);
    x1f = fminf(fmaxf(x1f, 0.f), 255.f);
    y0f = fminf(fmaxf(y0f, 0.f), 255.f);
    y1f = fminf(fmaxf(y1f, 0.f), 255.f);
    float wa = (x1f - xf) * (y1f - yf);
    float wb = (x1f - xf) * (yf - y0f);
    float wc = (xf - x0f) * (y1f - yf);
    float wd = (xf - x0f) * (yf - y0f);
    int xi0 = (int)x0f, xi1 = (int)x1f, yi0 = (int)y0f, yi1 = (int)y1f;
    size_t xb  = (size_t)b * 256 * 256;
    size_t a00 = xb + (size_t)yi0 * 256 + xi0;
    size_t a10 = xb + (size_t)yi1 * 256 + xi0;
    size_t a01 = xb + (size_t)yi0 * 256 + xi1;
    size_t a11 = xb + (size_t)yi1 * 256 + xi1;
    // nearest-upsample gather: loc_up[Y,X,c] == x_int[Y>>3, X>>3, c]
    size_t lb  = (size_t)b * 32 * 32 * 31;
    size_t l00 = lb + (size_t)((yi0 >> 3) * 32 + (xi0 >> 3)) * 31;
    size_t l10 = lb + (size_t)((yi1 >> 3) * 32 + (xi0 >> 3)) * 31;
    size_t l01 = lb + (size_t)((yi0 >> 3) * 32 + (xi1 >> 3)) * 31;
    size_t l11 = lb + (size_t)((yi1 >> 3) * 32 + (xi1 >> 3)) * 31;
    float r[4];
#pragma unroll
    for (int q = 0; q < 4; ++q) {
        int c = g * 4 + q;
        float Ia, Ib, Ic, Id;
        if (c == 0) {
            Ia = x[a00]; Ib = x[a10]; Ic = x[a01]; Id = x[a11];
        } else {
            int cc = c - 1;
            Ia = xi[l00 + cc]; Ib = xi[l10 + cc];
            Ic = xi[l01 + cc]; Id = xi[l11 + cc];
        }
        float val = wa * Ia + wb * Ib + wc * Ic + wd * Id;
        r[q] = val >= 0.f ? val : NEG_SLOPE * val;
    }
    float4* op = (float4*)(out + ((size_t)((b * 256 + Y) * 256 + X)) * 32 + g * 4);
    *op = make_float4(r[0], r[1], r[2], r[3]);
}

static inline size_t align256(size_t v) { return (v + 255) & ~(size_t)255; }

extern "C" void kernel_launch(void* const* d_in, const int* in_sizes, int n_in,
                              void* d_out, int out_size, void* d_ws, size_t ws_size,
                              hipStream_t stream) {
    const float* x    = (const float*)d_in[0];
    const float* c1k  = (const float*)d_in[1];
    const float* c1b  = (const float*)d_in[2];
    const float* c2k  = (const float*)d_in[3];
    const float* c2b  = (const float*)d_in[4];
    const float* c3k  = (const float*)d_in[5];
    const float* c3b  = (const float*)d_in[6];
    const float* bng  = (const float*)d_in[7];
    const float* bnb  = (const float*)d_in[8];
    const float* bnm  = (const float*)d_in[9];
    const float* bnv  = (const float*)d_in[10];
    const float* d1k  = (const float*)d_in[11];
    const float* d1b  = (const float*)d_in[12];
    const float* d2k  = (const float*)d_in[13];
    const float* d2b  = (const float*)d_in[14];
    const float* d3k  = (const float*)d_in[15];
    const float* d3b  = (const float*)d_in[16];
    float* out = (float*)d_out;

    // workspace carve-up
    char* ws = (char*)d_ws;
    size_t off = 0;
    float* p1 = (float*)(ws + off); off = align256(off + (size_t)BATCH*128*128*8 * 4);
    float* p2 = (float*)(ws + off); off = align256(off + (size_t)BATCH*64*64*16  * 4);
    float* xt = (float*)(ws + off); off = align256(off + (size_t)BATCH*32*32*31  * 4);
    _Float16* fh = (_Float16*)(ws + off); off = align256(off + (size_t)BATCH*FEAT * 2);
    _Float16* wp = (_Float16*)(ws + off); off = align256(off + (size_t)4*KSTEPS*32*16 * 2);
    float* f1p   = (float*)(ws + off); off = align256(off + (size_t)KCHUNKS*16*64 * 4);
    float* thetab= (float*)(ws + off); off = align256(off + (size_t)BATCH*6  * 4);
    (void)ws_size; (void)in_sizes; (void)n_in; (void)out_size;

    int n1 = BATCH * 128 * 128;
    conv1_pool<<<(n1 + 255) / 256, 256, 0, stream>>>(x, c1k, c1b, p1);
    int n2 = BATCH * 64 * 64;
    conv2_pool<<<(n2 + 255) / 256, 256, 0, stream>>>(p1, c2k, c2b, p2);
    int n3 = BATCH * 32 * 32;
    conv3_pool<<<(n3 + 255) / 256, 256, 0, stream>>>(p2, c3k, c3b, xt);
    int n4 = BATCH * FEAT;
    bn_cast<<<(n4 + 255) / 256, 256, 0, stream>>>(xt, bng, bnb, bnm, bnv, fh);
    int n5 = 4 * KSTEPS * 32;
    pack_w1<<<(n5 + 255) / 256, 256, 0, stream>>>(d1k, wp);
    dense1_wmma<<<dim3(KCHUNKS, 4), 32, 0, stream>>>(fh, wp, f1p);
    dense23<<<BATCH, 96, 0, stream>>>(f1p, d1b, d2k, d2b, d3k, d3b, thetab);
    int n7 = BATCH * 256 * 256 * 8;
    sampler<<<(n7 + 255) / 256, 256, 0, stream>>>(x, xt, thetab, out);
}
